// GraphSAGE_5153960755665
// MI455X (gfx1250) — matrix operand order repaired and checked by reading, not compiled, not run on hardware
//
#include <hip/hip_runtime.h>
#include <hip/hip_bf16.h>

// ---------------------------------------------------------------------------
// GraphSAGE 2-layer forward for MI455X (gfx1250, wave32, WMMA bf16).
// v3: 32x128 wave tile in the GEMM (2 row-blocks x 8 col-fragments) so each
// B fragment is reused by 2 WMMAs and each A fragment by 8 -> 0.625 b128
// loads per v_wmma (was 1.5), halving L2 traffic of the dominant GEMM.
// All GEMM A-operands pre-converted to row-major bf16; WMMA inner loop is
// pure b128 loads + v_wmma_f32_16x16x32_bf16.
// ---------------------------------------------------------------------------

#define DFEAT 256

typedef __attribute__((ext_vector_type(16))) __bf16        v16bf;
typedef __attribute__((ext_vector_type(8)))  float         v8f;
typedef __attribute__((ext_vector_type(2)))  float         v2f;
typedef __attribute__((ext_vector_type(4)))  unsigned int  v4u;

union BFrag { v16bf v; unsigned int u[8]; v4u q[2]; };

__device__ __forceinline__ unsigned int bfbits(float f) {
    unsigned int u = __builtin_bit_cast(unsigned int, f);
    u += 0x7FFFu + ((u >> 16) & 1u);          // round-to-nearest-even
    return u >> 16;
}
__device__ __forceinline__ unsigned int pack_bf2(float a, float b) {
    return bfbits(a) | (bfbits(b) << 16);
}

// ---------------------------------------------------------------------------
// Neighbor sampling + weight normalization. Deterministic column bijection
// j = (13*s + seed) & 63 stands in for the host-side random permutation.
// ---------------------------------------------------------------------------
template <int S>
__global__ __launch_bounds__(256) void sample_k(
    const int* __restrict__ par, int npar,
    const int* __restrict__ didx, const float* __restrict__ dval,
    int seed, int* __restrict__ outIdx, float* __restrict__ outW)
{
    int i = blockIdx.x * 256 + threadIdx.x;
    if (i >= npar) return;
    int id = par[i];
    const int*   ip = didx + (long)id * 64;
    const float* vp = dval + (long)id * 64;
    float vv[S];
    float s = 0.f;
#pragma unroll
    for (int k = 0; k < S; ++k) {
        int j = (k * 13 + seed) & 63;
        outIdx[i * S + k] = ip[j];
        float v = vp[j];
        vv[k] = v;
        s += v;
    }
    float inv = 1.f / (s + 1e-10f);
#pragma unroll
    for (int k = 0; k < S; ++k) outW[i * S + k] = vv[k] * inv;
}

// ---------------------------------------------------------------------------
// Pack a 256x256 fp32 weight matrix (row = K, col = N) into bf16 WMMA
// B-fragments, fragment-contiguous: frag f = kb*16+nb holds 256 dwords,
// dword index = lane*8 + j; lane covers N = nb*16 + (lane&15),
// K = kb*32 + 8*(lane>>4) + 2*j + (j>=4 ? 8 : 0) (+1 in high half of dword).
// ---------------------------------------------------------------------------
__global__ __launch_bounds__(256) void pack_w(
    const float* __restrict__ W, unsigned int* __restrict__ Wp)
{
    int t    = blockIdx.x * 256 + threadIdx.x;   // < 32768
    int f    = t >> 8;
    int r    = t & 255;
    int lane = r >> 3;
    int j    = r & 7;
    int kb   = f >> 4;
    int nb   = f & 15;
    int n    = nb * 16 + (lane & 15);
    int K    = kb * 32 + 8 * (lane >> 4) + 2 * j + ((j >> 2) << 3);
    Wp[t] = pack_bf2(W[(long)K * DFEAT + n], W[(long)(K + 1) * DFEAT + n]);
}

__global__ void bias_k(const float* __restrict__ a, const float* __restrict__ b,
                       float* __restrict__ o)
{
    int i = threadIdx.x;
    o[i] = a[i] + b[i];
}

// ---------------------------------------------------------------------------
// Gather fp32 rows and emit row-major bf16 (dword = 2 packed bf16).
// One block per node, thread t -> columns 2t, 2t+1.
// ---------------------------------------------------------------------------
__global__ __launch_bounds__(128) void gather_cvt(
    const float* __restrict__ feats, const int* __restrict__ idx,
    unsigned int* __restrict__ out)
{
    int n = blockIdx.x;
    int t = threadIdx.x;                       // 0..127
    int r = idx[n];
    v2f f = *(const v2f*)(feats + (long)r * DFEAT + 2 * t);
    out[(long)n * 128 + t] = pack_bf2(f.x, f.y);
}

// ---------------------------------------------------------------------------
// Weighted mean of gathered neighbor rows -> packed bf16 output.
// out[n, 2t:2t+2] = sum_s w[n,s] * feats[idx[n,s], 2t:2t+2]
// feats table (51 MB) is L2-resident.
// ---------------------------------------------------------------------------
__global__ __launch_bounds__(128) void wmean_g(
    const float* __restrict__ feats, const int* __restrict__ idx,
    const float* __restrict__ w, int S, unsigned int* __restrict__ out)
{
    int n = blockIdx.x;
    int t = threadIdx.x;
    float a0 = 0.f, a1 = 0.f;
    for (int s = 0; s < S; ++s) {
        int r = idx[n * S + s];
        if (s + 1 < S)  // pull next gathered row toward the caches early
            __builtin_prefetch(feats + (long)idx[n * S + s + 1] * DFEAT + 2 * t, 0, 1);
        float ww = w[n * S + s];
        v2f f = *(const v2f*)(feats + (long)r * DFEAT + 2 * t);
        a0 += ww * f.x;
        a1 += ww * f.y;
    }
    out[(long)n * 128 + t] = pack_bf2(a0, a1);
}

// Weighted mean over contiguous child rows (fp32 in) -> packed bf16 out.
__global__ __launch_bounds__(128) void wmean_c(
    const float* __restrict__ H, const float* __restrict__ w, int S,
    unsigned int* __restrict__ out)
{
    int n = blockIdx.x;
    int t = threadIdx.x;
    float a0 = 0.f, a1 = 0.f;
    for (int s = 0; s < S; ++s) {
        float ww = w[n * S + s];
        v2f f = *(const v2f*)(H + (long)(n * S + s) * DFEAT + 2 * t);
        a0 += ww * f.x;
        a1 += ww * f.y;
    }
    out[(long)n * 128 + t] = pack_bf2(a0, a1);
}

// ---------------------------------------------------------------------------
// Fused out = relu(X@Wx + M@Wn + bias), D = 256, X/M row-major bf16.
// One wave -> 32(M) x 128(N) tile: 2 row-blocks x 8 col-fragments,
// 16 v8f accumulators, 256 wmma/wave. Each B fragment feeds 2 WMMAs (both
// row-blocks); each A fragment feeds 8 (all col-fragments).
// Tile counts divide exactly: EXEC all-ones at every WMMA.
// Optionally writes a bf16 copy of the output (for feeding the next layer).
// ---------------------------------------------------------------------------
__global__ __launch_bounds__(256) void gemm_relu(
    const unsigned short* __restrict__ X, const unsigned short* __restrict__ M,
    const unsigned int* __restrict__ Wxp, const unsigned int* __restrict__ Wnp,
    const float* __restrict__ bias, float* __restrict__ out,
    unsigned short* __restrict__ out_bf, int nrows)
{
    (void)nrows;
    const int lane    = threadIdx.x & 31;
    const int wave    = threadIdx.x >> 5;
    const int gtile   = blockIdx.x * 8 + wave;
    const int colgrp  = gtile & 1;        // 2 x 128-col groups
    const int rowtile = gtile >> 1;
    const int r0      = rowtile * 32;
    const int khalf   = lane >> 4;
    const int l16     = lane & 15;

    const unsigned short* xrow0 = X + (long)(r0 + l16) * DFEAT;
    const unsigned short* xrow1 = X + (long)(r0 + 16 + l16) * DFEAT;
    const unsigned short* mrow0 = M + (long)(r0 + l16) * DFEAT;
    const unsigned short* mrow1 = M + (long)(r0 + 16 + l16) * DFEAT;

    v8f c[2][8];
#pragma unroll
    for (int rb = 0; rb < 2; ++rb)
#pragma unroll
        for (int t = 0; t < 8; ++t) c[rb][t] = v8f{};

#pragma unroll
    for (int kb = 0; kb < 8; ++kb) {
        const int koff = kb * 32 + 8 * khalf;
        // ---- A fragments from X: two contiguous 8-bf16 runs each ----
        BFrag ax0, ax1;
        ax0.q[0] = *(const v4u*)(xrow0 + koff);
        ax0.q[1] = *(const v4u*)(xrow0 + koff + 16);
        ax1.q[0] = *(const v4u*)(xrow1 + koff);
        ax1.q[1] = *(const v4u*)(xrow1 + koff + 16);
#pragma unroll
        for (int t = 0; t < 8; ++t) {
            BFrag b;
            const unsigned int* bp =
                Wxp + ((((kb << 4) + (colgrp << 3) + t) << 8) + (lane << 3));
            b.q[0] = *(const v4u*)bp;
            b.q[1] = *(const v4u*)(bp + 4);
            c[0][t] = __builtin_amdgcn_wmma_f32_16x16x32_bf16(
                false, ax0.v, false, b.v, (short)0, c[0][t], false, false);
            c[1][t] = __builtin_amdgcn_wmma_f32_16x16x32_bf16(
                false, ax1.v, false, b.v, (short)0, c[1][t], false, false);
        }
        // ---- A fragments from M ----
        BFrag am0, am1;
        am0.q[0] = *(const v4u*)(mrow0 + koff);
        am0.q[1] = *(const v4u*)(mrow0 + koff + 16);
        am1.q[0] = *(const v4u*)(mrow1 + koff);
        am1.q[1] = *(const v4u*)(mrow1 + koff + 16);
#pragma unroll
        for (int t = 0; t < 8; ++t) {
            BFrag b;
            const unsigned int* bp =
                Wnp + ((((kb << 4) + (colgrp << 3) + t) << 8) + (lane << 3));
            b.q[0] = *(const v4u*)bp;
            b.q[1] = *(const v4u*)(bp + 4);
            c[0][t] = __builtin_amdgcn_wmma_f32_16x16x32_bf16(
                false, am0.v, false, b.v, (short)0, c[0][t], false, false);
            c[1][t] = __builtin_amdgcn_wmma_f32_16x16x32_bf16(
                false, am1.v, false, b.v, (short)0, c[1][t], false, false);
        }
    }

    // ---- epilogue: bias + relu; C layout VGPR r: M = r + 8*khalf ----
#pragma unroll
    for (int rb = 0; rb < 2; ++rb) {
#pragma unroll
        for (int t = 0; t < 8; ++t) {
            int col = colgrp * 128 + t * 16 + l16;
            float bv = bias[col];
#pragma unroll
            for (int r = 0; r < 8; ++r) {
                int row = r0 + rb * 16 + r + 8 * khalf;
                float v = c[rb][t][r] + bv;
                v = v > 0.f ? v : 0.f;
                out[(long)row * DFEAT + col] = v;
                if (out_bf)
                    out_bf[(long)row * DFEAT + col] = (unsigned short)bfbits(v);
            }
        }
    }
}

// ---------------------------------------------------------------------------
extern "C" void kernel_launch(void* const* d_in, const int* in_sizes, int n_in,
                              void* d_out, int out_size, void* d_ws, size_t ws_size,
                              hipStream_t stream)
{
    (void)in_sizes; (void)n_in; (void)out_size; (void)ws_size;
    const int*   ids   = (const int*)d_in[0];
    const int*   didx  = (const int*)d_in[1];
    const float* dval  = (const float*)d_in[2];
    const float* feats = (const float*)d_in[3];
    const float* Wx1 = (const float*)d_in[4];
    const float* bx1 = (const float*)d_in[5];
    const float* Wn1 = (const float*)d_in[6];
    const float* bn1 = (const float*)d_in[7];
    const float* Wx2 = (const float*)d_in[8];
    const float* bx2 = (const float*)d_in[9];
    const float* Wn2 = (const float*)d_in[10];
    const float* bn2 = (const float*)d_in[11];

    constexpr int BATCH = 1024, S1 = 25, S2 = 10;
    constexpr int N1 = BATCH * S1;        // 25600
    constexpr int N2 = N1 * S2;           // 256000

    char* p = (char*)d_ws;
    auto take = [&](size_t bytes) -> char* {
        char* r = p;
        p += (bytes + 255) & ~(size_t)255;
        return r;
    };
    int*            cur1 = (int*)take((size_t)N1 * 4);
    float*          w1   = (float*)take((size_t)N1 * 4);
    int*            cur2 = (int*)take((size_t)N2 * 4);
    float*          w2   = (float*)take((size_t)N2 * 4);
    unsigned short* Xg1  = (unsigned short*)take((size_t)N1 * DFEAT * 2);    // bf16
    unsigned short* Xg0  = (unsigned short*)take((size_t)BATCH * DFEAT * 2); // bf16
    unsigned short* M1b  = (unsigned short*)take((size_t)N1 * DFEAT * 2);    // bf16
    unsigned short* M0b  = (unsigned short*)take((size_t)BATCH * DFEAT * 2); // bf16
    float*          H1   = (float*)take((size_t)N1 * DFEAT * 4);             // fp32
    float*          H0   = (float*)take((size_t)BATCH * DFEAT * 4);          // fp32
    unsigned short* H0b  = (unsigned short*)take((size_t)BATCH * DFEAT * 2); // bf16
    unsigned short* M0pb = (unsigned short*)take((size_t)BATCH * DFEAT * 2); // bf16
    unsigned*       Wp1x = (unsigned*)take(32768u * 4);
    unsigned*       Wp1n = (unsigned*)take(32768u * 4);
    unsigned*       Wp2x = (unsigned*)take(32768u * 4);
    unsigned*       Wp2n = (unsigned*)take(32768u * 4);
    float*          b1   = (float*)take(DFEAT * 4);
    float*          b2   = (float*)take(DFEAT * 4);

    // 1) sampling (level 1 then level 2)
    sample_k<S1><<<BATCH / 256, 256, 0, stream>>>(ids,  BATCH, didx, dval, 7,  cur1, w1);
    sample_k<S2><<<N1 / 256,    256, 0, stream>>>(cur1, N1,    didx, dval, 14, cur2, w2);

    // 2) pack weights to bf16 WMMA fragments; combine biases
    pack_w<<<128, 256, 0, stream>>>(Wx1, Wp1x);
    pack_w<<<128, 256, 0, stream>>>(Wn1, Wp1n);
    pack_w<<<128, 256, 0, stream>>>(Wx2, Wp2x);
    pack_w<<<128, 256, 0, stream>>>(Wn2, Wp2n);
    bias_k<<<1, 256, 0, stream>>>(bx1, bn1, b1);
    bias_k<<<1, 256, 0, stream>>>(bx2, bn2, b2);

    // 3) gather+convert X operands; weighted neighbor means -> bf16
    gather_cvt<<<N1,    128, 0, stream>>>(feats, cur1, (unsigned int*)Xg1);
    gather_cvt<<<BATCH, 128, 0, stream>>>(feats, ids,  (unsigned int*)Xg0);
    wmean_g<<<N1,    128, 0, stream>>>(feats, cur2, w2, S2, (unsigned int*)M1b);
    wmean_g<<<BATCH, 128, 0, stream>>>(feats, cur1, w1, S1, (unsigned int*)M0b);

    // 4) layer-1 fused GEMMs: H = relu(X@Wx1 + M@Wn1 + b1)
    //    wave-tiles = (rows/32) * 2 colgrps; 8 waves per block, exact division
    gemm_relu<<<N1 / 32 * 2 / 8, 256, 0, stream>>>(Xg1, M1b, Wp1x, Wp1n, b1,
                                                   H1, nullptr, N1);
    gemm_relu<<<BATCH / 32 * 2 / 8, 256, 0, stream>>>(Xg0, M0b, Wp1x, Wp1n, b1,
                                                      H0, H0b, BATCH);

    // 5) layer-2: mean over contiguous H1 children, then final fused GEMM
    wmean_c<<<BATCH, 128, 0, stream>>>(H1, w1, S1, (unsigned int*)M0pb);
    gemm_relu<<<BATCH / 32 * 2 / 8, 256, 0, stream>>>(H0b, M0pb, Wp2x, Wp2n, b2,
                                                      (float*)d_out, nullptr, BATCH);
}
